// React_attention_84464826843818
// MI455X (gfx1250) — matrix operand order repaired and checked by reading, not compiled, not run on hardware
//
#include <hip/hip_runtime.h>
#include <stdint.h>

#define BATCH 8
#define S_ENC 2048
#define S_DEC 1024
#define DIN   512
#define UNITS 512

typedef __attribute__((ext_vector_type(16))) __bf16         v16bf;
typedef __attribute__((ext_vector_type(8)))  float          v8f;
typedef __attribute__((ext_vector_type(8)))  unsigned int   v8u;
typedef __attribute__((ext_vector_type(8)))  unsigned short us8;
typedef __attribute__((ext_vector_type(4)))  int            v4i;

// typed address-space pointers matching the async-LDS builtin signature
typedef __attribute__((address_space(1))) v4i glob_v4i;
typedef __attribute__((address_space(3))) v4i lds_v4i;

// gfx1250 async global->LDS path (ASYNCcnt-tracked), with sync fallback
#if defined(__has_builtin)
#  if __has_builtin(__builtin_amdgcn_global_load_async_to_lds_b128) && \
      __has_builtin(__builtin_amdgcn_s_wait_asynccnt)
#    define HAVE_ASYNC_LDS 1
#  endif
#endif
#ifndef HAVE_ASYNC_LDS
#  define HAVE_ASYNC_LDS 0
#endif

__device__ __forceinline__ unsigned short bf16_bits(float f) {
  return __builtin_bit_cast(unsigned short, (__bf16)f);
}
__device__ __forceinline__ unsigned int bf16x2_bits(float lo, float hi) {
  return (unsigned int)bf16_bits(lo) | ((unsigned int)bf16_bits(hi) << 16);
}

__device__ __forceinline__ v8f wmma_bf16(v8u a, v8u b, v8f c) {
  return __builtin_amdgcn_wmma_f32_16x16x32_bf16(
      false, __builtin_bit_cast(v16bf, a),
      false, __builtin_bit_cast(v16bf, b),
      (short)0, c, false, false);
}

// ---------------------------------------------------------------------------
// Projection GEMM: Out[M,512](bf16) = X[M,512](f32) @ W[512,512](f32)
// grid = (M/64, 512/64), block = 128 threads (4 waves).
// ---------------------------------------------------------------------------
__global__ __launch_bounds__(128) void proj_bf16_kernel(
    const float* __restrict__ X, const float* __restrict__ W,
    unsigned short* __restrict__ Out)
{
  __shared__ unsigned short Xs [64][34];   // [row][k], +2 pad
  __shared__ unsigned short Wts[64][34];   // transposed: [n][k]

  const int tid   = threadIdx.x;
  const int wave  = tid >> 5;
  const int lane  = tid & 31;
  const int lhalf = lane >> 4;
  const int l16   = lane & 15;
  const int row0  = blockIdx.x * 64;
  const int col0  = blockIdx.y * 64;

  v8f acc[4];
  #pragma unroll
  for (int j = 0; j < 4; ++j)
    #pragma unroll
    for (int v = 0; v < 8; ++v) acc[j][v] = 0.0f;

  for (int kk = 0; kk < DIN; kk += 32) {
    // stage X tile 64x32: f32 -> bf16, packed b32 LDS stores
    #pragma unroll
    for (int r = 0; r < 4; ++r) {
      int i  = tid + 128 * r;
      int xr = i >> 3;
      int xc = (i & 7) << 2;
      const float4 f = *(const float4*)&X[(size_t)(row0 + xr) * DIN + kk + xc];
      *(unsigned int*)&Xs[xr][xc + 0] = bf16x2_bits(f.x, f.y);
      *(unsigned int*)&Xs[xr][xc + 2] = bf16x2_bits(f.z, f.w);
    }
    // stage W tile 32x64 transposed -> [n][k]
    #pragma unroll
    for (int r = 0; r < 4; ++r) {
      int i  = tid + 128 * r;
      int wr = i >> 4;
      int wc = (i & 15) << 2;
      const float4 f = *(const float4*)&W[(size_t)(kk + wr) * UNITS + col0 + wc];
      Wts[wc + 0][wr] = bf16_bits(f.x);
      Wts[wc + 1][wr] = bf16_bits(f.y);
      Wts[wc + 2][wr] = bf16_bits(f.z);
      Wts[wc + 3][wr] = bf16_bits(f.w);
    }
    __syncthreads();

    v8u av;
    {
      const int arow = 16 * wave + l16;
      #pragma unroll
      for (int v = 0; v < 8; ++v) {
        int k = ((v < 4) ? 2 * v : 16 + 2 * (v - 4)) + (lhalf ? 8 : 0);
        av[v] = *(const unsigned int*)&Xs[arow][k];
      }
    }
    #pragma unroll
    for (int j = 0; j < 4; ++j) {
      const int n = 16 * j + l16;
      v8u bv;
      #pragma unroll
      for (int v = 0; v < 8; ++v) {
        int k = ((v < 4) ? 2 * v : 16 + 2 * (v - 4)) + (lhalf ? 8 : 0);
        bv[v] = *(const unsigned int*)&Wts[n][k];
      }
      acc[j] = wmma_bf16(av, bv, acc[j]);
    }
    __syncthreads();
  }

  #pragma unroll
  for (int j = 0; j < 4; ++j)
    #pragma unroll
    for (int v = 0; v < 8; ++v) {
      int r = 16 * wave + (lhalf ? 8 : 0) + v;
      int c = col0 + 16 * j + l16;
      Out[(size_t)(row0 + r) * UNITS + c] = bf16_bits(acc[j][v]);
    }
}

// ---------------------------------------------------------------------------
// Fused flash attention. One block per (batch, 16-query tile), 256 thr / 8 waves.
// K/V tiles double-buffered in LDS; K streamed with async global->LDS (b128),
// V staged with a packed transpose. Q fragments live in registers.
// ---------------------------------------------------------------------------
__global__ __launch_bounds__(256) void attn_kernel(
    const unsigned short* __restrict__ Qb,
    const unsigned short* __restrict__ Kb,
    const unsigned short* __restrict__ Vb,
    float* __restrict__ Out)
{
  __shared__ unsigned short Ks [2][32][DIN];    // 64 KB  [key][d]
  __shared__ unsigned short Vts[2][UNITS][34];  // 68 KB  [u][key]
  __shared__ float          Sf [16][32];        // score reduction tile
  __shared__ unsigned short Pb [16][34];        // exp(scores) bf16
  __shared__ float m_sh[16], l_sh[16], corr_sh[16];

  const int tid   = threadIdx.x;
  const int wave  = tid >> 5;
  const int lane  = tid & 31;
  const int lhalf = lane >> 4;
  const int l16   = lane & 15;

  const int q0 = blockIdx.x * 16;
  const int b  = blockIdx.y;
  const size_t qrow0 = (size_t)b * S_DEC + q0;
  const size_t krow0 = (size_t)b * S_ENC;
  const int NB = S_ENC / 32;

  // Resident Q A-fragments: this wave only ever needs d-chunk [64*wave, +64)
  v8u aQ[2];
  #pragma unroll
  for (int ks = 0; ks < 2; ++ks)
    #pragma unroll
    for (int v = 0; v < 8; ++v) {
      int k = 64 * wave + 32 * ks + ((v < 4) ? 2 * v : 16 + 2 * (v - 4)) + (lhalf ? 8 : 0);
      aQ[ks][v] = *(const unsigned int*)&Qb[(qrow0 + l16) * UNITS + k];
    }

  if (tid < 16) { m_sh[tid] = -__builtin_inff(); l_sh[tid] = 0.0f; }

  v8f accO[4];
  #pragma unroll
  for (int j = 0; j < 4; ++j)
    #pragma unroll
    for (int v = 0; v < 8; ++v) accO[j][v] = 0.0f;

  // --- staging helpers -----------------------------------------------------
  auto stageK = [&](int nb, int kb) {
    #pragma unroll
    for (int r = 0; r < 8; ++r) {
      int i  = tid + 256 * r;
      int kr = i >> 6;
      int kc = (i & 63) << 3;
      const unsigned short* g = &Kb[(krow0 + kb + kr) * UNITS + kc];
      unsigned short*       l = &Ks[nb][kr][kc];
#if HAVE_ASYNC_LDS
      __builtin_amdgcn_global_load_async_to_lds_b128(
          (glob_v4i*)g, (lds_v4i*)l, 0, 0);
#else
      *(uint4*)l = *(const uint4*)g;
#endif
    }
  };
  auto stageV = [&](int nb, int kb) {  // transpose: Vts[u][key], packed key-pairs
    #pragma unroll
    for (int r = 0; r < 4; ++r) {
      int i  = tid + 256 * r;
      int kp = i >> 6;                 // key pair 0..15
      int vu = (i & 63) << 3;
      us8 e = *(const us8*)&Vb[(krow0 + kb + 2 * kp + 0) * UNITS + vu];
      us8 o = *(const us8*)&Vb[(krow0 + kb + 2 * kp + 1) * UNITS + vu];
      #pragma unroll
      for (int j = 0; j < 8; ++j)
        *(unsigned int*)&Vts[nb][vu + j][2 * kp] =
            (unsigned int)e[j] | ((unsigned int)o[j] << 16);
    }
  };
  auto prefetchKV = [&](int kb) {      // next-next tile -> global_prefetch
    const char* pk = (const char*)&Kb[(krow0 + kb) * UNITS];
    const char* pv = (const char*)&Vb[(krow0 + kb) * UNITS];
    __builtin_prefetch(pk + tid * 128, 0, 0);
    __builtin_prefetch(pv + tid * 128, 0, 0);
  };

  // prologue: stage block 0 into buffer 0
  stageK(0, 0);
  stageV(0, 0);

  for (int kbi = 0; kbi < NB; ++kbi) {
    const int buf = kbi & 1;

    // zero score tile for this iteration (softmax consumed it last iter, B2 passed)
    (&Sf[0][0])[tid]       = 0.0f;
    (&Sf[0][0])[tid + 256] = 0.0f;

    // stage next block into the other buffer; prefetch two blocks ahead
    if (kbi + 1 < NB) {
      stageK(buf ^ 1, (kbi + 1) * 32);
      stageV(buf ^ 1, (kbi + 1) * 32);
      if (kbi + 2 < NB) prefetchKV((kbi + 2) * 32);
    }
#if HAVE_ASYNC_LDS
    // only the 8 just-issued async ops may remain outstanding -> current buffer ready
    if (kbi + 1 < NB) __builtin_amdgcn_s_wait_asynccnt(8);
    else              __builtin_amdgcn_s_wait_asynccnt(0);
#endif
    __syncthreads();  // Btop: staging of `buf` visible, Sf zero visible

    // ---- scores: wave handles d-chunk [64*wave, +64), keys [kb, kb+32)
    {
      v8f accS[2];
      #pragma unroll
      for (int f = 0; f < 2; ++f)
        #pragma unroll
        for (int v = 0; v < 8; ++v) accS[f][v] = 0.0f;

      #pragma unroll
      for (int ks = 0; ks < 2; ++ks) {
        const int d0 = 64 * wave + 32 * ks;
        #pragma unroll
        for (int f = 0; f < 2; ++f) {
          const int key = 16 * f + l16;
          v8u bv;
          #pragma unroll
          for (int v = 0; v < 8; ++v) {
            int k = d0 + ((v < 4) ? 2 * v : 16 + 2 * (v - 4)) + (lhalf ? 8 : 0);
            bv[v] = *(const unsigned int*)&Ks[buf][key][k];
          }
          accS[f] = wmma_bf16(aQ[ks], bv, accS[f]);
        }
      }
      // cross-wave partial-score reduction via LDS float atomics (ds_add_f32)
      #pragma unroll
      for (int f = 0; f < 2; ++f)
        #pragma unroll
        for (int v = 0; v < 8; ++v) {
          int mrow = (lhalf ? 8 : 0) + v;
          int ncol = 16 * f + l16;
          atomicAdd(&Sf[mrow][ncol], accS[f][v]);
        }
    }
    __syncthreads();  // B1

    // ---- online softmax update, one thread per query row
    if (tid < 16) {
      float mOld = m_sh[tid];
      float mx   = mOld;
      #pragma unroll
      for (int n = 0; n < 32; ++n) mx = fmaxf(mx, Sf[tid][n]);
      float c    = __expf(mOld - mx);
      float psum = 0.0f;
      #pragma unroll
      for (int n = 0; n < 32; ++n) {
        float p = __expf(Sf[tid][n] - mx);
        psum += p;
        Pb[tid][n] = bf16_bits(p);
      }
      m_sh[tid]    = mx;
      l_sh[tid]    = l_sh[tid] * c + psum;
      corr_sh[tid] = c;
    }
    __syncthreads();  // B2

    // ---- O = O*corr + P @ V ; wave owns u-columns [64*wave, +64)
    {
      v8u av;
      #pragma unroll
      for (int v = 0; v < 8; ++v) {
        int k = ((v < 4) ? 2 * v : 16 + 2 * (v - 4)) + (lhalf ? 8 : 0);
        av[v] = *(const unsigned int*)&Pb[l16][k];
      }
      #pragma unroll
      for (int j = 0; j < 4; ++j) {
        #pragma unroll
        for (int v = 0; v < 8; ++v) {
          int mrow = (lhalf ? 8 : 0) + v;
          accO[j][v] *= corr_sh[mrow];
        }
        const int u = 64 * wave + 16 * j + l16;
        v8u bv;
        #pragma unroll
        for (int v = 0; v < 8; ++v) {
          int k = ((v < 4) ? 2 * v : 16 + 2 * (v - 4)) + (lhalf ? 8 : 0);
          bv[v] = *(const unsigned int*)&Vts[buf][u][k];
        }
        accO[j] = wmma_bf16(av, bv, accO[j]);
      }
    }
  }

  // ---- normalize by running sum, write f32 output
  #pragma unroll
  for (int j = 0; j < 4; ++j)
    #pragma unroll
    for (int v = 0; v < 8; ++v) {
      int mrow = (lhalf ? 8 : 0) + v;
      int u    = 64 * wave + 16 * j + l16;
      Out[(qrow0 + mrow) * UNITS + u] = accO[j][v] / l_sh[mrow];
    }
}

// ---------------------------------------------------------------------------
extern "C" void kernel_launch(void* const* d_in, const int* in_sizes, int n_in,
                              void* d_out, int out_size, void* d_ws, size_t ws_size,
                              hipStream_t stream) {
  (void)in_sizes; (void)n_in; (void)out_size; (void)ws_size;

  const float* encoding = (const float*)d_in[0];  // [8,2048,512]
  const float* decoding = (const float*)d_in[1];  // [8,1024,512]
  const float* Wq       = (const float*)d_in[2];  // [512,512]
  const float* Wk       = (const float*)d_in[3];
  const float* Wv       = (const float*)d_in[4];
  float*       out      = (float*)d_out;          // [8,1024,512] f32

  // bf16 Q/K/V in workspace: 8 MB + 16 MB + 16 MB = 40 MB
  unsigned short* Qb = (unsigned short*)d_ws;
  unsigned short* Kb = Qb + (size_t)BATCH * S_DEC * UNITS;
  unsigned short* Vb = Kb + (size_t)BATCH * S_ENC * UNITS;

  proj_bf16_kernel<<<dim3((BATCH * S_DEC) / 64, UNITS / 64), 128, 0, stream>>>(decoding, Wq, Qb);
  proj_bf16_kernel<<<dim3((BATCH * S_ENC) / 64, UNITS / 64), 128, 0, stream>>>(encoding, Wk, Kb);
  proj_bf16_kernel<<<dim3((BATCH * S_ENC) / 64, UNITS / 64), 128, 0, stream>>>(encoding, Wv, Vb);

  attn_kernel<<<dim3(S_DEC / 16, BATCH), 256, 0, stream>>>(Qb, Kb, Vb, out);
}
